// GAT_36971078484045
// MI455X (gfx1250) — matrix operand order repaired and checked by reading, not compiled, not run on hardware
//
#include <hip/hip_runtime.h>
#include <hip/hip_bf16.h>

// ---------------------------------------------------------------------------
// GAT (2-layer) for MI455X / gfx1250, wave32.
//  - Dense projections: v_wmma_f32_16x16x32_f16 (one wave per 16x16 tile).
//  - Edge softmax/aggregation: L2-resident float atomics. All hot buffers
//    (proj1 12.8MB, agg1 12.8MB, att1 51.2MB, edge_index 12.8MB) fit in the
//    192MB L2, so the edge passes are L2-bandwidth / atomic bound, not HBM.
//  - Attention coefficients are materialized once per (edge,head) and the
//    scatter pass gathers proj[src] as b128 — no redundant transcendentals.
// ---------------------------------------------------------------------------

typedef _Float16 half_t;
typedef _Float16 v16h __attribute__((ext_vector_type(16)));
typedef float    v8f  __attribute__((ext_vector_type(8)));

#define N_NODES 50000
#define N_EDGES 1600000

__device__ __forceinline__ float lrelu02(float x) { return x > 0.f ? x : 0.2f * x; }

// Monotonic float -> uint encoding so atomicMax(u32) implements float max.
__device__ __forceinline__ unsigned fenc(float f) {
    unsigned u = __float_as_uint(f);
    return (u & 0x80000000u) ? ~u : (u | 0x80000000u);
}
__device__ __forceinline__ float fdec(unsigned u) {
    return __uint_as_float((u & 0x80000000u) ? (u & 0x7FFFFFFFu) : ~u);
}
#define ENC_NEG_INF 0x007FFFFFu   // fenc(-inf)

// ---------------------------------------------------------------------------
// f16 GEMM via WMMA: out[M x ncols] = A[M x K] * B^T  (B stored [ncols_pad x K])
// One wave per 16x16 output tile. K % 32 == 0, M % 16 == 0.
// Fragment layouts per CDNA5 ISA 7.12.2 (wave32).
// ---------------------------------------------------------------------------
__global__ __launch_bounds__(32) void k_gemm_f16_wmma(
    float* __restrict__ out, const half_t* __restrict__ A,
    const half_t* __restrict__ B, int K, int ncols)
{
    const int lane = threadIdx.x;
    const int m0   = blockIdx.x * 16;
    const int n0   = blockIdx.y * 16;
    const int mlo  = lane & 15;
    const bool lo  = lane < 16;

    v8f c = {};
    const half_t* Arow = A + (size_t)(m0 + mlo) * K;                 // A row = node
    const half_t* Brow = B + (size_t)(n0 + mlo) * K;                 // B row = out-feature
    const int a0 = lo ? 0 : 8;    // 16-bit A 16x32: K for halves 0..7
    const int a1 = lo ? 16 : 24;  //                 K for halves 8..15

    for (int kb = 0; kb < K; kb += 32) {
        v16h a, b;
#pragma unroll
        for (int h = 0; h < 8; ++h)  a[h]     = Arow[kb + a0 + h];
#pragma unroll
        for (int h = 0; h < 8; ++h)  a[8 + h] = Arow[kb + a1 + h];
#pragma unroll
        for (int h = 0; h < 8; ++h)  b[h]     = Brow[kb + a0 + h];
#pragma unroll
        for (int h = 0; h < 8; ++h)  b[8 + h] = Brow[kb + a1 + h];
        c = __builtin_amdgcn_wmma_f32_16x16x32_f16(false, a, false, b,
                                                   (short)0, c, false, false);
    }

    const int col = n0 + mlo;
    if (col < ncols) {
        const int r0 = lo ? 0 : 8;   // f32 C/D 16x16: VGPR r -> row M = r0 + r
#pragma unroll
        for (int r = 0; r < 8; ++r)
            out[(size_t)(m0 + r0 + r) * ncols + col] = c[r];
    }
}

// ---------------------------------------------------------------------------
// Utility / elementwise kernels
// ---------------------------------------------------------------------------
__global__ void k_f32_to_f16(half_t* __restrict__ dst, const float* __restrict__ src, int n) {
    int i = blockIdx.x * blockDim.x + threadIdx.x;
    if (i < n) dst[i] = (half_t)src[i];
}

// Pad W2 [7 x 64] -> f16 [16 x 64] (rows 7..15 zero) so one 16-wide WMMA tile works.
__global__ void k_pad_w2(half_t* __restrict__ dst, const float* __restrict__ W2) {
    int i = blockIdx.x * blockDim.x + threadIdx.x;
    if (i < 16 * 64) {
        int c = i >> 6, k = i & 63;
        dst[i] = (half_t)(c < 7 ? W2[c * 64 + k] : 0.f);
    }
}

// Per-node attention logits: s_src[n,h] = <proj[n,h,:], a_src[h,:]> (same for trg).
__global__ void k_attn_prep(float* __restrict__ s_src, float* __restrict__ s_trg,
                            const float* __restrict__ proj,
                            const float* __restrict__ a_src,
                            const float* __restrict__ a_trg, int H, int F) {
    int i = blockIdx.x * blockDim.x + threadIdx.x;
    if (i >= N_NODES * H) return;
    int n = i / H, h = i % H;
    const float* p  = proj + (size_t)n * (H * F) + h * F;
    const float* as = a_src + h * F;
    const float* at = a_trg + h * F;
    float ss = 0.f, st = 0.f;
    for (int f = 0; f < F; ++f) { ss += p[f] * as[f]; st += p[f] * at[f]; }
    s_src[i] = ss; s_trg[i] = st;
}

// Zero aggregation buffer [N,C]; init segment max/sum [N,H].
__global__ void k_init(float* __restrict__ agg, unsigned* __restrict__ m,
                       float* __restrict__ sum, int C, int H) {
    long long i = (long long)blockIdx.x * blockDim.x + threadIdx.x;
    if (i < (long long)N_NODES * C) agg[i] = 0.f;
    if (i < (long long)N_NODES * H) { m[i] = ENC_NEG_INF; sum[i] = 0.f; }
}

// Pass 1 over edges: segment max of leaky-relu scores (per target node, per head).
__global__ void k_edge_max(unsigned* __restrict__ m,
                           const float* __restrict__ s_src, const float* __restrict__ s_trg,
                           const int* __restrict__ ei, int H) {
    long long i = (long long)blockIdx.x * blockDim.x + threadIdx.x;
    if (i >= (long long)N_EDGES * H) return;
    int e = (int)(i / H), h = (int)(i % H);
    int s = ei[e], t = ei[N_EDGES + e];
    float sc = lrelu02(s_src[s * H + h] + s_trg[t * H + h]);
    atomicMax(&m[t * H + h], fenc(sc));
}

// Pass 2 (fast path): p = exp(score - max); segment-sum it AND store to att[e,h]
// (coalesced store: i == e*H+h).
__global__ void k_edge_sum_store(float* __restrict__ sum, float* __restrict__ att,
                                 const unsigned* __restrict__ m,
                                 const float* __restrict__ s_src, const float* __restrict__ s_trg,
                                 const int* __restrict__ ei, int H) {
    long long i = (long long)blockIdx.x * blockDim.x + threadIdx.x;
    if (i >= (long long)N_EDGES * H) return;
    int e = (int)(i / H), h = (int)(i % H);
    int s = ei[e], t = ei[N_EDGES + e];
    float sc = lrelu02(s_src[s * H + h] + s_trg[t * H + h]);
    float p  = expf(sc - fdec(m[t * H + h]));
    att[i] = p;
    atomicAdd(&sum[t * H + h], p);
}

// Pass 2b (fast path): normalize att in place (coalesced read/write, one gather).
__global__ void k_att_norm(float* __restrict__ att, const float* __restrict__ sum,
                           const int* __restrict__ ei, int H) {
    long long i = (long long)blockIdx.x * blockDim.x + threadIdx.x;
    if (i >= (long long)N_EDGES * H) return;
    int e = (int)(i / H), h = (int)(i % H);
    int t = ei[N_EDGES + e];
    att[i] = att[i] / (sum[t * H + h] + 1e-16f);
}

// Pass 3 (fast path, layer 1): one thread per (edge, 4-feature group).
// b128 gather of proj[src], 4 float atomic adds; att pre-normalized.
__global__ void k_edge_agg_v4(float* __restrict__ agg, const float* __restrict__ proj,
                              const float* __restrict__ att, const int* __restrict__ ei) {
    long long i = (long long)blockIdx.x * blockDim.x + threadIdx.x;
    if (i >= (long long)N_EDGES * 16) return;        // 64 feats / 4 per thread
    int e  = (int)(i >> 4);
    int f0 = (int)(i & 15) * 4;                      // group start; h = f0>>3 (F=8)
    int s = ei[e], t = ei[N_EDGES + e];
    float a = att[(size_t)e * 8 + (f0 >> 3)];
    const float4 p = *(const float4*)(proj + (size_t)s * 64 + f0);
    float* dst = agg + (size_t)t * 64 + f0;
    atomicAdd(dst + 0, p.x * a);
    atomicAdd(dst + 1, p.y * a);
    atomicAdd(dst + 2, p.z * a);
    atomicAdd(dst + 3, p.w * a);
}

// Pass 3 (fast path, generic scalar; used for layer 2: H=1, F=7).
__global__ void k_edge_agg_pre(float* __restrict__ agg, const float* __restrict__ proj,
                               const float* __restrict__ att, const int* __restrict__ ei,
                               int H, int F) {
    const int C = H * F;
    long long i = (long long)blockIdx.x * blockDim.x + threadIdx.x;
    if (i >= (long long)N_EDGES * C) return;
    int e = (int)(i / C), f = (int)(i % C);
    int s = ei[e], t = ei[N_EDGES + e];
    float a = att[(size_t)e * H + f / F];
    atomicAdd(&agg[(size_t)t * C + f], proj[(size_t)s * C + f] * a);
}

// ---- Fallback (small workspace): recompute att per feature ----
__global__ void k_edge_sum(float* __restrict__ sum, const unsigned* __restrict__ m,
                           const float* __restrict__ s_src, const float* __restrict__ s_trg,
                           const int* __restrict__ ei, int H) {
    long long i = (long long)blockIdx.x * blockDim.x + threadIdx.x;
    if (i >= (long long)N_EDGES * H) return;
    int e = (int)(i / H), h = (int)(i % H);
    int s = ei[e], t = ei[N_EDGES + e];
    float sc = lrelu02(s_src[s * H + h] + s_trg[t * H + h]);
    atomicAdd(&sum[t * H + h], expf(sc - fdec(m[t * H + h])));
}
__global__ void k_edge_agg(float* __restrict__ agg, const float* __restrict__ proj,
                           const float* __restrict__ s_src, const float* __restrict__ s_trg,
                           const unsigned* __restrict__ m, const float* __restrict__ sum,
                           const int* __restrict__ ei, int H, int F) {
    const int C = H * F;
    long long i = (long long)blockIdx.x * blockDim.x + threadIdx.x;
    if (i >= (long long)N_EDGES * C) return;
    int e = (int)(i / C), f = (int)(i % C);
    int h = f / F;
    int s = ei[e], t = ei[N_EDGES + e];
    float sc  = lrelu02(s_src[s * H + h] + s_trg[t * H + h]);
    float a   = expf(sc - fdec(m[t * H + h])) / (sum[t * H + h] + 1e-16f);
    atomicAdd(&agg[(size_t)t * C + f], proj[(size_t)s * C + f] * a);
}

// h = elu(agg1 + b1), cast to f16 for the layer-2 WMMA GEMM.
__global__ void k_elu_f16(half_t* __restrict__ hh, const float* __restrict__ agg,
                          const float* __restrict__ b1) {
    int i = blockIdx.x * blockDim.x + threadIdx.x;
    if (i >= N_NODES * 64) return;
    float v = agg[i] + b1[i & 63];
    hh[i] = (half_t)(v > 0.f ? v : (expf(v) - 1.f));
}

// out[n,:] = softmax(agg2[n,:] + b2) over 7 classes.
__global__ void k_final(float* __restrict__ out, const float* __restrict__ agg2,
                        const float* __restrict__ b2) {
    int n = blockIdx.x * blockDim.x + threadIdx.x;
    if (n >= N_NODES) return;
    float z[7], mx = -3.4e38f;
#pragma unroll
    for (int c = 0; c < 7; ++c) { z[c] = agg2[n * 7 + c] + b2[c]; mx = fmaxf(mx, z[c]); }
    float s = 0.f;
#pragma unroll
    for (int c = 0; c < 7; ++c) { z[c] = expf(z[c] - mx); s += z[c]; }
#pragma unroll
    for (int c = 0; c < 7; ++c) out[n * 7 + c] = z[c] / s;
}

// ---------------------------------------------------------------------------
extern "C" void kernel_launch(void* const* d_in, const int* in_sizes, int n_in,
                              void* d_out, int out_size, void* d_ws, size_t ws_size,
                              hipStream_t stream) {
    (void)in_sizes; (void)n_in; (void)out_size;

    const float* x      = (const float*)d_in[0];
    const int*   ei     = (const int*)  d_in[1];
    const float* W1     = (const float*)d_in[2];
    const float* a_src1 = (const float*)d_in[3];
    const float* a_trg1 = (const float*)d_in[4];
    const float* b1     = (const float*)d_in[5];
    const float* W2     = (const float*)d_in[6];
    const float* a_src2 = (const float*)d_in[7];
    const float* a_trg2 = (const float*)d_in[8];
    const float* b2     = (const float*)d_in[9];

    char* ws = (char*)d_ws;
    size_t off = 0;
    auto take = [&](size_t bytes) -> char* {
        char* p = ws + off;
        off += (bytes + 255) & ~(size_t)255;
        return p;
    };

    half_t*   xh    = (half_t*)  take((size_t)N_NODES * 128 * sizeof(half_t));
    half_t*   w1h   = (half_t*)  take(64 * 128 * sizeof(half_t));
    float*    proj1 = (float*)   take((size_t)N_NODES * 64 * sizeof(float));
    float*    ssrc1 = (float*)   take((size_t)N_NODES * 8 * sizeof(float));
    float*    strg1 = (float*)   take((size_t)N_NODES * 8 * sizeof(float));
    unsigned* m1    = (unsigned*)take((size_t)N_NODES * 8 * sizeof(unsigned));
    float*    sum1  = (float*)   take((size_t)N_NODES * 8 * sizeof(float));
    float*    agg1  = (float*)   take((size_t)N_NODES * 64 * sizeof(float));
    half_t*   hh    = (half_t*)  take((size_t)N_NODES * 64 * sizeof(half_t));
    half_t*   w2h   = (half_t*)  take(16 * 64 * sizeof(half_t));
    float*    proj2 = (float*)   take((size_t)N_NODES * 7 * sizeof(float));
    float*    ssrc2 = (float*)   take((size_t)N_NODES * sizeof(float));
    float*    strg2 = (float*)   take((size_t)N_NODES * sizeof(float));
    unsigned* m2    = (unsigned*)take((size_t)N_NODES * sizeof(unsigned));
    float*    sum2  = (float*)   take((size_t)N_NODES * sizeof(float));
    float*    agg2  = (float*)   take((size_t)N_NODES * 7 * sizeof(float));
    // Optional (fast-path) attention buffers appended last:
    float*    att1  = (float*)   take((size_t)N_EDGES * 8 * sizeof(float));
    float*    att2  = (float*)   take((size_t)N_EDGES * sizeof(float));
    const bool fast = (off <= ws_size);   // constant per harness -> deterministic

    const int B = 256;
    auto nb = [](long long total, int b) { return (unsigned)((total + b - 1) / b); };

    // ---- Layer 1 ----
    k_f32_to_f16<<<nb((long long)N_NODES * 128, B), B, 0, stream>>>(xh, x, N_NODES * 128);
    k_f32_to_f16<<<nb(64 * 128, B), B, 0, stream>>>(w1h, W1, 64 * 128);
    k_gemm_f16_wmma<<<dim3(N_NODES / 16, 4), 32, 0, stream>>>(proj1, xh, w1h, 128, 64);
    k_attn_prep<<<nb((long long)N_NODES * 8, B), B, 0, stream>>>(ssrc1, strg1, proj1,
                                                                 a_src1, a_trg1, 8, 8);
    k_init<<<nb((long long)N_NODES * 64, B), B, 0, stream>>>(agg1, m1, sum1, 64, 8);
    k_edge_max<<<nb((long long)N_EDGES * 8, B), B, 0, stream>>>(m1, ssrc1, strg1, ei, 8);
    if (fast) {
        k_edge_sum_store<<<nb((long long)N_EDGES * 8, B), B, 0, stream>>>(sum1, att1, m1,
                                                                          ssrc1, strg1, ei, 8);
        k_att_norm<<<nb((long long)N_EDGES * 8, B), B, 0, stream>>>(att1, sum1, ei, 8);
        k_edge_agg_v4<<<nb((long long)N_EDGES * 16, B), B, 0, stream>>>(agg1, proj1, att1, ei);
    } else {
        k_edge_sum<<<nb((long long)N_EDGES * 8, B), B, 0, stream>>>(sum1, m1, ssrc1, strg1, ei, 8);
        k_edge_agg<<<nb((long long)N_EDGES * 64, B), B, 0, stream>>>(agg1, proj1, ssrc1, strg1,
                                                                     m1, sum1, ei, 8, 8);
    }
    k_elu_f16<<<nb((long long)N_NODES * 64, B), B, 0, stream>>>(hh, agg1, b1);

    // ---- Layer 2 ----
    k_pad_w2<<<nb(16 * 64, B), B, 0, stream>>>(w2h, W2);
    k_gemm_f16_wmma<<<dim3(N_NODES / 16, 1), 32, 0, stream>>>(proj2, hh, w2h, 64, 7);
    k_attn_prep<<<nb(N_NODES, B), B, 0, stream>>>(ssrc2, strg2, proj2, a_src2, a_trg2, 1, 7);
    k_init<<<nb((long long)N_NODES * 7, B), B, 0, stream>>>(agg2, m2, sum2, 7, 1);
    k_edge_max<<<nb(N_EDGES, B), B, 0, stream>>>(m2, ssrc2, strg2, ei, 1);
    if (fast) {
        k_edge_sum_store<<<nb(N_EDGES, B), B, 0, stream>>>(sum2, att2, m2, ssrc2, strg2, ei, 1);
        k_att_norm<<<nb(N_EDGES, B), B, 0, stream>>>(att2, sum2, ei, 1);
        k_edge_agg_pre<<<nb((long long)N_EDGES * 7, B), B, 0, stream>>>(agg2, proj2, att2, ei, 1, 7);
    } else {
        k_edge_sum<<<nb(N_EDGES, B), B, 0, stream>>>(sum2, m2, ssrc2, strg2, ei, 1);
        k_edge_agg<<<nb((long long)N_EDGES * 7, B), B, 0, stream>>>(agg2, proj2, ssrc2, strg2,
                                                                    m2, sum2, ei, 1, 7);
    }
    k_final<<<nb(N_NODES, B), B, 0, stream>>>((float*)d_out, agg2, b2);
}
